// SimpleLSTM_45569603010723
// MI455X (gfx1250) — compile-verified, matrix operand order checked by reading, not run on hardware
//
#include <hip/hip_runtime.h>
#include <math.h>

// Problem constants (from reference)
constexpr int Bb   = 64;
constexpr int Ss   = 512;
constexpr int Ii   = 512;
constexpr int Oo   = 512;
constexpr int Hh   = 1024;
constexpr int COMB = Ii + Hh;   // 1536
constexpr int G4   = 4 * Hh;    // 4096

typedef __attribute__((ext_vector_type(16))) __bf16 v16bf;
typedef __attribute__((ext_vector_type(8)))  float  v8f;

// ---------- helpers ----------
__device__ __forceinline__ unsigned short f2bf(float f) {
    unsigned u = __builtin_bit_cast(unsigned, f);
    u += 0x7FFFu + ((u >> 16) & 1u);   // round-to-nearest-even
    return (unsigned short)(u >> 16);
}

union FragU {
    uint4          q[2];
    unsigned short us[16];
    v16bf          v;
};

// 16-bit WMMA fragment: two contiguous 8-element bf16 chunks (16B each)
__device__ __forceinline__ v16bf load_bf16_frag(const unsigned short* p) {
    FragU u;
    u.q[0] = *reinterpret_cast<const uint4*>(p);
    u.q[1] = *reinterpret_cast<const uint4*>(p + 16);
    return u.v;
}

__device__ __forceinline__ float sigmoidf_(float x) {
    return 1.0f / (1.0f + __expf(-x));
}

// fast tanh via hardware exp; clamp keeps exp finite
__device__ __forceinline__ float tanhf_(float x) {
    x = fminf(15.0f, fmaxf(-15.0f, x));
    float e = __expf(2.0f * x);
    return (e - 1.0f) / (e + 1.0f);
}

// ---------- one-time conversions ----------
__global__ __launch_bounds__(256)
void convert_gate_weights(const float* __restrict__ Wf, const float* __restrict__ Wi,
                          const float* __restrict__ Wc, const float* __restrict__ Wo,
                          unsigned short* __restrict__ Wg) {
    const size_t total = (size_t)G4 * COMB;
    for (size_t i = (size_t)blockIdx.x * blockDim.x + threadIdx.x; i < total;
         i += (size_t)gridDim.x * blockDim.x) {
        size_t row = i / COMB;
        size_t k   = i - row * COMB;
        int    g   = (int)(row / Hh);
        size_t r   = row - (size_t)g * Hh;
        const float* src = (g == 0) ? Wf : (g == 1) ? Wi : (g == 2) ? Wc : Wo;
        Wg[i] = f2bf(src[r * COMB + k]);
    }
}

__global__ __launch_bounds__(256)
void convert_wy(const float* __restrict__ Wy, unsigned short* __restrict__ Wyb) {
    const size_t total = (size_t)Oo * Hh;
    for (size_t i = (size_t)blockIdx.x * blockDim.x + threadIdx.x; i < total;
         i += (size_t)gridDim.x * blockDim.x) {
        Wyb[i] = f2bf(Wy[i]);
    }
}

// x is scan-invariant: convert the whole [B,S,I] tensor to bf16 once so the
// 512 recurrent gate kernels do zero conversion VALU work in their K loops.
__global__ __launch_bounds__(256)
void convert_x(const float* __restrict__ x, unsigned short* __restrict__ xbf) {
    const size_t total = (size_t)Bb * Ss * Ii;
    for (size_t i = (size_t)blockIdx.x * blockDim.x + threadIdx.x; i < total;
         i += (size_t)gridDim.x * blockDim.x) {
        xbf[i] = f2bf(x[i]);
    }
}

__global__ __launch_bounds__(256)
void init_state(const float* __restrict__ h0, const float* __restrict__ c0,
                unsigned short* __restrict__ hist0, float* __restrict__ c) {
    const size_t total = (size_t)Bb * Hh;
    for (size_t i = (size_t)blockIdx.x * blockDim.x + threadIdx.x; i < total;
         i += (size_t)gridDim.x * blockDim.x) {
        hist0[i] = f2bf(h0[i]);
        c[i]     = c0[i];
    }
}

// ---------- fused gate GEMM + LSTM cell update (one time step) ----------
// grid = (8, 4), block = 256 (8 waves). Each wave owns a 16x16 (B x Hcols)
// tile and computes ALL FOUR gates for it (4 fp32 WMMA accumulators), so the
// sigmoid/tanh gating and the c/h update happen in-register in the epilogue.
// Loads are consumed directly by the WMMAs; unroll-by-8 gives the scheduler
// rotating registers and a deep window to pipeline loads ahead of the waits.
__global__ __launch_bounds__(256)
void lstm_gate_step(const unsigned short* __restrict__ xbf,    // [B,S,I] bf16
                    const unsigned short* __restrict__ Wg,      // [4H,COMB] bf16
                    const float* __restrict__ bfv, const float* __restrict__ biv,
                    const float* __restrict__ bcv, const float* __restrict__ bov,
                    const unsigned short* __restrict__ h_prev,  // [B,H] bf16
                    unsigned short* __restrict__ h_out,         // [B,H] bf16
                    float* __restrict__ h_f32,                  // [B,H] fp32
                    float* __restrict__ c,                      // [B,H] fp32
                    int t) {
    const int lane = threadIdx.x & 31;
    const int wave = threadIdx.x >> 5;
    const int lm   = lane & 15;
    const int half = lane >> 4;            // K-half selector per ISA layout
    const int kA   = half * 8;

    const int n0 = blockIdx.x * 128 + wave * 16;   // H-column base of tile
    const int m0 = blockIdx.y * 16;                // batch-row base of tile

    // A-matrix rows for this lane (lane = M)
    const int mA = m0 + lm;
    const unsigned short* xrow = xbf + ((size_t)mA * Ss + t) * Ii + kA;
    const unsigned short* hrow = h_prev + (size_t)mA * Hh + kA;

    // B-matrix rows: Wg row (gate*H + n0 + lane) holds column n of B (lane = N)
    const unsigned short* Brow[4];
#pragma unroll
    for (int g = 0; g < 4; ++g)
        Brow[g] = Wg + (size_t)(g * Hh + n0 + lm) * COMB + kA;

    // bias loads issued early so their latency hides under the GEMM
    const int   ncol = n0 + lm;
    const float Bf = bfv[ncol], Bi = biv[ncol], Bc = bcv[ncol], Bo = bov[ncol];

    v8f acc[4];
#pragma unroll
    for (int g = 0; g < 4; ++g) acc[g] = v8f{};

    // K over x-part
#pragma unroll 8
    for (int k0 = 0; k0 < Ii; k0 += 32) {
        v16bf a = load_bf16_frag(xrow + k0);
#pragma unroll
        for (int g = 0; g < 4; ++g) {
            v16bf b = load_bf16_frag(Brow[g] + k0);
            acc[g] = __builtin_amdgcn_wmma_f32_16x16x32_bf16(
                false, a, false, b, (short)0, acc[g], false, false);
        }
    }
    // K over h-part (bf16 recurrent state)
#pragma unroll 8
    for (int k0 = Ii; k0 < COMB; k0 += 32) {
        v16bf a = load_bf16_frag(hrow + (k0 - Ii));
#pragma unroll
        for (int g = 0; g < 4; ++g) {
            v16bf b = load_bf16_frag(Brow[g] + k0);
            acc[g] = __builtin_amdgcn_wmma_f32_16x16x32_bf16(
                false, a, false, b, (short)0, acc[g], false, false);
        }
    }

    // Epilogue: C/D layout — lane = N, VGPR v = M (lanes 16..31: M = 8+v)
    const int mbase = m0 + half * 8;
#pragma unroll
    for (int v = 0; v < 8; ++v) {
        float f  = sigmoidf_(acc[0][v] + Bf);
        float i  = sigmoidf_(acc[1][v] + Bi);
        float ct = tanhf_(acc[2][v] + Bc);
        float o  = sigmoidf_(acc[3][v] + Bo);
        size_t idx = (size_t)(mbase + v) * Hh + ncol;
        float cn = f * c[idx] + i * ct;
        c[idx]   = cn;
        float hn = o * tanhf_(cn);
        h_out[idx] = f2bf(hn);
        h_f32[idx] = hn;
    }
}

// ---------- final output projection: [S*B, H] @ [H, O] in one big GEMM ----------
// Each wave computes a 16M x 64N strip (1 A-frag reused across 4 WMMAs/iter).
__global__ __launch_bounds__(256)
void lstm_y_gemm(const unsigned short* __restrict__ Hm,   // [S*B, H] bf16
                 const unsigned short* __restrict__ Wyb,  // [O, H]   bf16
                 const float* __restrict__ by,
                 float* __restrict__ out) {               // [B, S, O] fp32
    const int lane = threadIdx.x & 31;
    const int wave = threadIdx.x >> 5;
    const int lm   = lane & 15;
    const int half = lane >> 4;
    const int kA   = half * 8;

    const int tile = blockIdx.x * 8 + wave;
    const int mt   = tile >> 3;          // 2048 M-tiles
    const int ng   = tile & 7;           // 8 N-groups of 64
    const int m0   = mt * 16;
    const int n0   = ng * 64;

    const unsigned short* Arow = Hm + (size_t)(m0 + lm) * Hh + kA;
    const unsigned short* Brow[4];
#pragma unroll
    for (int j = 0; j < 4; ++j)
        Brow[j] = Wyb + (size_t)(n0 + j * 16 + lm) * Hh + kA;

    v8f acc[4];
#pragma unroll
    for (int j = 0; j < 4; ++j) acc[j] = v8f{};

#pragma unroll 8
    for (int k0 = 0; k0 < Hh; k0 += 32) {
        v16bf a = load_bf16_frag(Arow + k0);
#pragma unroll
        for (int j = 0; j < 4; ++j) {
            v16bf b = load_bf16_frag(Brow[j] + k0);
            acc[j] = __builtin_amdgcn_wmma_f32_16x16x32_bf16(
                false, a, false, b, (short)0, acc[j], false, false);
        }
    }

#pragma unroll
    for (int j = 0; j < 4; ++j) {
        const int   ncol = n0 + j * 16 + lm;
        const float bb   = by[ncol];
#pragma unroll
        for (int v = 0; v < 8; ++v) {
            int r = m0 + half * 8 + v;        // row index in (s*B + b) order
            int b = r & (Bb - 1);
            int s = r >> 6;                    // r / B
            out[((size_t)b * Ss + s) * Oo + ncol] = acc[j][v] + bb;
        }
    }
}

// ---------- tuple tail: h_t, c_t in fp32 ----------
__global__ __launch_bounds__(256)
void finalize_state(const float* __restrict__ h_f32, const float* __restrict__ c,
                    float* __restrict__ out_h, float* __restrict__ out_c) {
    const size_t total = (size_t)Bb * Hh;
    for (size_t i = (size_t)blockIdx.x * blockDim.x + threadIdx.x; i < total;
         i += (size_t)gridDim.x * blockDim.x) {
        out_h[i] = h_f32[i];
        out_c[i] = c[i];
    }
}

// ---------- launcher ----------
extern "C" void kernel_launch(void* const* d_in, const int* in_sizes, int n_in,
                              void* d_out, int out_size, void* d_ws, size_t ws_size,
                              hipStream_t stream) {
    (void)in_sizes; (void)n_in; (void)out_size; (void)ws_size;

    const float* x   = (const float*)d_in[0];
    const float* h0  = (const float*)d_in[1];
    const float* c0  = (const float*)d_in[2];
    const float* Wf  = (const float*)d_in[3];
    const float* bfv = (const float*)d_in[4];
    const float* Wi  = (const float*)d_in[5];
    const float* biv = (const float*)d_in[6];
    const float* Wc  = (const float*)d_in[7];
    const float* bcv = (const float*)d_in[8];
    const float* Wo  = (const float*)d_in[9];
    const float* bov = (const float*)d_in[10];
    const float* Wy  = (const float*)d_in[11];
    const float* by  = (const float*)d_in[12];

    float* out = (float*)d_out;
    float* out_h = out + (size_t)Bb * Ss * Oo;
    float* out_c = out_h + (size_t)Bb * Hh;

    // workspace layout
    char* ws = (char*)d_ws;
    unsigned short* Wg_bf = (unsigned short*)ws;                          // 12.6 MB
    ws += (size_t)G4 * COMB * sizeof(unsigned short);
    unsigned short* Wy_bf = (unsigned short*)ws;                          // 1 MB
    ws += (size_t)Oo * Hh * sizeof(unsigned short);
    unsigned short* x_bf = (unsigned short*)ws;                           // 32 MB
    ws += (size_t)Bb * Ss * Ii * sizeof(unsigned short);
    float* c_st = (float*)ws;                                             // 256 KB
    ws += (size_t)Bb * Hh * sizeof(float);
    float* h_f32 = (float*)ws;                                            // 256 KB
    ws += (size_t)Bb * Hh * sizeof(float);
    unsigned short* hist = (unsigned short*)ws;                           // 67.2 MB
    // hist: (S+1) slabs of [B, H] bf16; slab 0 = h0

    // one-time conversions + state init
    convert_gate_weights<<<2048, 256, 0, stream>>>(Wf, Wi, Wc, Wo, Wg_bf);
    convert_wy<<<512, 256, 0, stream>>>(Wy, Wy_bf);
    convert_x<<<4096, 256, 0, stream>>>(x, x_bf);
    init_state<<<256, 256, 0, stream>>>(h0, c0, hist, c_st);

    // sequential scan: fused gate GEMM + cell update per step
    const size_t slab = (size_t)Bb * Hh;
    for (int t = 0; t < Ss; ++t) {
        lstm_gate_step<<<dim3(8, 4), 256, 0, stream>>>(
            x_bf, Wg_bf, bfv, biv, bcv, bov,
            hist + (size_t)t * slab,        // h_{t}
            hist + (size_t)(t + 1) * slab,  // h_{t+1}
            h_f32, c_st, t);
    }

    // output projection over the whole history in one GEMM
    lstm_y_gemm<<<2048, 256, 0, stream>>>(hist + slab, Wy_bf, by, out);

    finalize_state<<<256, 256, 0, stream>>>(h_f32, c_st, out_h, out_c);
}
